// GINLayer_59459527246446
// MI455X (gfx1250) — matrix operand order repaired and checked by reading, not compiled, not run on hardware
//
#include <hip/hip_runtime.h>
#include <hip/hip_bf16.h>

#define NN 50000
#define NE 600000
#define DD 128
#define MTILES (NN / 16)   // 3125

typedef __attribute__((ext_vector_type(16))) __bf16 v16bf;
typedef __attribute__((ext_vector_type(8)))  __bf16 v8bf;
typedef __attribute__((ext_vector_type(4)))  __bf16 v4bf;
typedef __attribute__((ext_vector_type(8)))  float  v8f;

// ---------------- init: agg = x (fp32 copy, float4) ----------------
__global__ __launch_bounds__(256) void k_copy(const float* __restrict__ x,
                                              float* __restrict__ agg) {
  unsigned gid = blockIdx.x * blockDim.x + threadIdx.x;
  if (gid >= (unsigned)(NN * DD / 4)) return;
  ((float4*)agg)[gid] = ((const float4*)x)[gid];
}

// ---------------- zero BN accumulators ----------------
__global__ void k_zero(float* __restrict__ sums) {
  sums[threadIdx.x] = 0.0f;   // 256 floats: [0:128)=sum, [128:256)=sumsq
}

// ---------------- edge scatter: agg[dst] += x[src] ----------------
// one wave per edge: 32 lanes x float4 = full 128-wide row
__global__ __launch_bounds__(256) void k_scatter(const float* __restrict__ x,
                                                 const long long* __restrict__ ei,
                                                 float* __restrict__ agg) {
  unsigned gid = blockIdx.x * blockDim.x + threadIdx.x;
  if (gid >= (unsigned)NE * 32u) return;
  unsigned e  = gid >> 5;
  unsigned c4 = (gid & 31u) << 2;
  long long src = ei[e];
  long long dst = ei[(size_t)NE + e];
  float4 v = *(const float4*)(x + (size_t)src * DD + c4);
  float* p = agg + (size_t)dst * DD + c4;
  atomicAdd(p + 0, v.x);
  atomicAdd(p + 1, v.y);
  atomicAdd(p + 2, v.z);
  atomicAdd(p + 3, v.w);
}

// ---------------- fp32 -> bf16 (row-major) ----------------
__global__ __launch_bounds__(256) void k_tobf16(const float* __restrict__ in,
                                                __bf16* __restrict__ out) {
  unsigned gid = blockIdx.x * blockDim.x + threadIdx.x;
  if (gid >= (unsigned)(NN * DD / 4)) return;
  float4 v = ((const float4*)in)[gid];
  v4bf o = { (__bf16)v.x, (__bf16)v.y, (__bf16)v.z, (__bf16)v.w };
  *(v4bf*)(out + (size_t)gid * 4) = o;
}

// ---------------- swizzle W (fp32 row-major [k][n]) into WMMA B-fragments ----------
// frag index = ((ntile*4 + ks)*32 + lane)*16 + e
//   lane<16 : K = ks*32 + (e<8 ? e : 16+e-8),     col = lane
//   lane>=16: K = ks*32 + 8 + (e<8 ? e : 16+e-8), col = lane-16
__global__ __launch_bounds__(256) void k_wprep(const float* __restrict__ W,
                                               __bf16* __restrict__ Wfrag) {
  unsigned gid = blockIdx.x * blockDim.x + threadIdx.x;
  if (gid >= 16384u) return;
  unsigned e     = gid & 15u;
  unsigned lane  = (gid >> 4) & 31u;
  unsigned ks    = (gid >> 9) & 3u;
  unsigned ntile = gid >> 11;
  unsigned n = ntile * 16 + (lane & 15u);
  unsigned k = ks * 32 + (lane >> 4) * 8 + (e & 7u) + (e >> 3) * 16;
  Wfrag[gid] = (__bf16)W[k * DD + n];
}

// ---------------- WMMA GEMM -------------------------------------------------
// One M-tile (16 rows) per WAVE; A fragments (4 x v16bf) held in registers and
// reused across all 8 N-tiles => 32 WMMAs per wave, A loaded once from global.
// RELU_BF16==1: out = relu(acc + bias) -> bf16   (layer 1)
// RELU_BF16==0: out = acc + bias       -> fp32   (layer 2)
template <int RELU_BF16>
__global__ __launch_bounds__(256) void k_gemm_t(const __bf16* __restrict__ A,
                                                const __bf16* __restrict__ Bfrag,
                                                const float* __restrict__ bias,
                                                float* __restrict__ outF,
                                                __bf16* __restrict__ outH) {
  const int wave  = threadIdx.x >> 5;
  const int lane  = threadIdx.x & 31;
  const int mtile = blockIdx.x * 8 + wave;
  if (mtile >= MTILES) return;
  const int half = lane >> 4;
  const int r    = lane & 15;
  const int row0 = mtile * 16;

  // Load A fragments once (two contiguous 16B chunks per lane per K-step).
  const __bf16* arow = A + (size_t)(row0 + r) * DD + (half ? 8 : 0);
  union { v16bf v; v8bf h[2]; } a[4];
#pragma unroll
  for (int ks = 0; ks < 4; ++ks) {
    a[ks].h[0] = *(const v8bf*)(arow + ks * 32);
    a[ks].h[1] = *(const v8bf*)(arow + ks * 32 + 16);
  }

  const v16bf* bbase = (const v16bf*)Bfrag + lane;
#pragma unroll
  for (int nt = 0; nt < 8; ++nt) {
    v8f c = {};
#pragma unroll
    for (int ks = 0; ks < 4; ++ks) {
      v16bf b = bbase[(nt * 4 + ks) * 32];
      c = __builtin_amdgcn_wmma_f32_16x16x32_bf16(false, a[ks].v, false, b,
                                                  (short)0, c, false, false);
    }
    const int col = nt * 16 + r;
    const float bv = bias[col];
#pragma unroll
    for (int v = 0; v < 8; ++v) {
      const int row = row0 + half * 8 + v;
      float val = c[v] + bv;
      if (RELU_BF16) {
        val = fmaxf(val, 0.0f);
        outH[(size_t)row * DD + col] = (__bf16)val;
      } else {
        outF[(size_t)row * DD + col] = val;
      }
    }
  }
}

// ---------------- column-wise sum / sumsq over h ----------------
__global__ __launch_bounds__(256) void k_colstats(const float* __restrict__ h,
                                                  float* __restrict__ sums) {
  int c    = threadIdx.x & 127;
  int rsub = threadIdx.x >> 7;          // 0 or 1
  int rowBase = blockIdx.x * 256;
  float s = 0.0f, s2 = 0.0f;
  for (int i = rsub; i < 256; i += 2) {
    int row = rowBase + i;
    if (row < NN) {
      float v = h[(size_t)row * DD + c];
      s += v;
      s2 += v * v;
    }
  }
  atomicAdd(&sums[c], s);
  atomicAdd(&sums[128 + c], s2);
}

// ---------------- fold BN stats into per-column scale/shift ----------------
__global__ void k_bnprep(const float* __restrict__ sums,
                         const float* __restrict__ gamma,
                         const float* __restrict__ beta,
                         float* __restrict__ ss) {
  int c = threadIdx.x;
  if (c < 128) {
    const float invN = 1.0f / (float)NN;
    float mean = sums[c] * invN;
    float var  = sums[128 + c] * invN - mean * mean;
    float sc   = gamma[c] * rsqrtf(var + 1e-5f);
    ss[c]       = sc;
    ss[128 + c] = beta[c] - mean * sc;
  }
}

// ---------------- out = h*scale + shift + x ----------------
__global__ __launch_bounds__(256) void k_final(const float* __restrict__ h,
                                               const float* __restrict__ x,
                                               const float* __restrict__ ss,
                                               float* __restrict__ out) {
  unsigned gid = blockIdx.x * blockDim.x + threadIdx.x;
  if (gid >= (unsigned)(NN * DD / 4)) return;
  unsigned c4 = (gid & 31u) * 4;
  float4 hv = ((const float4*)h)[gid];
  float4 xv = ((const float4*)x)[gid];
  float4 sc = *(const float4*)(ss + c4);
  float4 sh = *(const float4*)(ss + 128 + c4);
  float4 o;
  o.x = hv.x * sc.x + sh.x + xv.x;
  o.y = hv.y * sc.y + sh.y + xv.y;
  o.z = hv.z * sc.z + sh.z + xv.z;
  o.w = hv.w * sc.w + sh.w + xv.w;
  ((float4*)out)[gid] = o;
}

extern "C" void kernel_launch(void* const* d_in, const int* in_sizes, int n_in,
                              void* d_out, int out_size, void* d_ws, size_t ws_size,
                              hipStream_t stream) {
  const float*     x     = (const float*)d_in[0];
  const long long* ei    = (const long long*)d_in[1];   // int64 edge_index [2][NE]
  const float*     W1    = (const float*)d_in[2];
  const float*     b1    = (const float*)d_in[3];
  const float*     W2    = (const float*)d_in[4];
  const float*     b2    = (const float*)d_in[5];
  const float*     gamma = (const float*)d_in[6];
  const float*     beta  = (const float*)d_in[7];
  float* out = (float*)d_out;

  // workspace layout
  char* ws = (char*)d_ws;
  float*  agg  = (float*)ws;                         // NN*DD fp32  (25.6 MB)
  float*  hbuf = agg + (size_t)NN * DD;              // NN*DD fp32  (25.6 MB)
  __bf16* zb   = (__bf16*)(hbuf + (size_t)NN * DD);  // NN*DD bf16  (12.8 MB)
  __bf16* h1b  = zb + (size_t)NN * DD;               // NN*DD bf16  (12.8 MB)
  __bf16* w1f  = h1b + (size_t)NN * DD;              // 16384 bf16 (32 KB)
  __bf16* w2f  = w1f + 16384;                        // 16384 bf16 (32 KB)
  float*  sums = (float*)(w2f + 16384);              // 256 fp32
  float*  ss   = sums + 256;                         // 256 fp32

  const unsigned n4 = NN * DD / 4;                   // 1,600,000
  const unsigned vecBlocks = (n4 + 255) / 256;       // 6250
  const unsigned gemmBlocks = (MTILES + 7) / 8;      // 391

  // 1) agg = x ; zero BN accumulators
  k_copy<<<vecBlocks, 256, 0, stream>>>(x, agg);
  k_zero<<<1, 256, 0, stream>>>(sums);

  // 2) edge scatter-add
  k_scatter<<<(NE * 32 + 255) / 256, 256, 0, stream>>>(x, ei, agg);

  // 3) z -> bf16
  k_tobf16<<<vecBlocks, 256, 0, stream>>>(agg, zb);

  // 4) weight fragment prep
  k_wprep<<<64, 256, 0, stream>>>(W1, w1f);
  k_wprep<<<64, 256, 0, stream>>>(W2, w2f);

  // 5) h1 = relu(z @ W1 + b1)  (bf16 out)
  k_gemm_t<1><<<gemmBlocks, 256, 0, stream>>>(zb, w1f, b1, nullptr, h1b);

  // 6) h = h1 @ W2 + b2  (fp32 out)
  k_gemm_t<0><<<gemmBlocks, 256, 0, stream>>>(h1b, w2f, b2, hbuf, nullptr);

  // 7) BN column stats
  k_colstats<<<(NN + 255) / 256, 256, 0, stream>>>(hbuf, sums);

  // 8) fold into scale/shift
  k_bnprep<<<1, 128, 0, stream>>>(sums, gamma, beta, ss);

  // 9) out = h*scale + shift + x
  k_final<<<vecBlocks, 256, 0, stream>>>(hbuf, x, ss, out);
}
